// GraphEmbedderSACN_13786845020225
// MI455X (gfx1250) — matrix operand order repaired and checked by reading, not compiled, not run on hardware
//
#include <hip/hip_runtime.h>
#include <hip/hip_bf16.h>

// ---------------------------------------------------------------------------
// GraphEmbedderSACN for MI455X (gfx1250, wave32).
// GEMMs use v_wmma_f32_16x16x32_bf16 (dense bf16 WMMA, f32 accum).
// Scatter/BN/gather phases are HBM/L2 bound (~4-5 GB traffic @ 23.3 TB/s).
// GCN biases are mathematically cancelled by the following BatchNorm.
// ---------------------------------------------------------------------------

typedef __attribute__((ext_vector_type(16))) __bf16 v16bf;
typedef __attribute__((ext_vector_type(8)))  __bf16 v8bf;
typedef __attribute__((ext_vector_type(8)))  float  v8f;

#define N_ENT   100000
#define N_REL   200
#define DIM     256
#define EFULL   (N_ENT * 16)
#define EBATCH  200000
#define K1      128
#define K3PAD   224
#define BN_EPS  1e-5f

// ---------------- utility kernels ----------------

__global__ void k_zero(float* __restrict__ p, size_t n) {
  size_t i = (size_t)blockIdx.x * blockDim.x + threadIdx.x;
  size_t stride = (size_t)gridDim.x * blockDim.x;
  for (; i < n; i += stride) p[i] = 0.0f;
}

// f32 [rows, kin] row-major -> bf16 [rows, kout] row-major, zero-padded in K
__global__ void k_cvt_pad(const float* __restrict__ in, __bf16* __restrict__ out,
                          int rows, int kin, int kout) {
  size_t total = (size_t)rows * kout;
  size_t i = (size_t)blockIdx.x * blockDim.x + threadIdx.x;
  size_t stride = (size_t)gridDim.x * blockDim.x;
  for (; i < total; i += stride) {
    int k = (int)(i % kout);
    size_t r = i / kout;
    float v = (k < kin) ? in[r * (size_t)kin + k] : 0.0f;
    out[i] = (__bf16)v;
  }
}

// W f32 [K, N] -> WT bf16 [N, Kpad]   (B^T copy so WMMA B-fragments load contiguously)
__global__ void k_transpose_cvt(const float* __restrict__ w, __bf16* __restrict__ wt,
                                int K, int N, int Kpad) {
  size_t total = (size_t)N * Kpad;
  size_t i = (size_t)blockIdx.x * blockDim.x + threadIdx.x;
  size_t stride = (size_t)gridDim.x * blockDim.x;
  for (; i < total; i += stride) {
    int k = (int)(i % Kpad);
    size_t n = i / Kpad;
    float v = (k < K) ? w[(size_t)k * N + n] : 0.0f;
    wt[i] = (__bf16)v;
  }
}

// ---------------- bf16 WMMA GEMM ----------------
// A bf16 [M, K] row-major, BT bf16 [N=256, K] row-major, C f32 [M, 256].
// One wave computes a 16x64 output strip: 1 A fragment x 4 B fragments.
// M % 16 == 0, K % 32 == 0 guaranteed by padding.
__global__ void k_gemm_wmma(const __bf16* __restrict__ A, const __bf16* __restrict__ BT,
                            float* __restrict__ C, int M, int K) {
  const int lane = threadIdx.x & 31;
  const int wave = threadIdx.x >> 5;
  const int wavetile = blockIdx.x * (blockDim.x >> 5) + wave;
  const int tiles_m = M >> 4;
  const int tm = wavetile >> 2;   // 16-row tile
  const int tg = wavetile & 3;    // 64-col group
  if (tm >= tiles_m) return;      // wave-uniform

  const int half = lane >> 4;                 // 0 or 1
  const int mrow = (tm << 4) + (lane & 15);
  // A layout: lane holds one row, 16 contiguous K; lane>=16 -> K base +16
  const __bf16* ap = A + (size_t)mrow * K + (half << 4);
  // B layout: lane holds one column, 16 contiguous K; lane>=16 -> K base +16
  const __bf16* bp = BT + (size_t)((tg << 6) + (lane & 15)) * K + (half << 4);
  const size_t bstep = (size_t)16 * K;

  v8f c0 = {}, c1 = {}, c2 = {}, c3 = {};
  for (int k = 0; k < K; k += 32) {
    v8bf alo = *(const v8bf*)(ap + k);
    v8bf ahi = *(const v8bf*)(ap + k + 8);
    v16bf a;
#pragma unroll
    for (int i = 0; i < 8; ++i) { a[i] = alo[i]; a[8 + i] = ahi[i]; }
    v16bf b0 = *(const v16bf*)(bp + k);
    v16bf b1 = *(const v16bf*)(bp + bstep + k);
    v16bf b2 = *(const v16bf*)(bp + 2 * bstep + k);
    v16bf b3 = *(const v16bf*)(bp + 3 * bstep + k);
    c0 = __builtin_amdgcn_wmma_f32_16x16x32_bf16(false, a, false, b0, (short)0, c0, false, false);
    c1 = __builtin_amdgcn_wmma_f32_16x16x32_bf16(false, a, false, b1, (short)0, c1, false, false);
    c2 = __builtin_amdgcn_wmma_f32_16x16x32_bf16(false, a, false, b2, (short)0, c2, false, false);
    c3 = __builtin_amdgcn_wmma_f32_16x16x32_bf16(false, a, false, b3, (short)0, c3, false, false);
  }
  // C layout: VGPR r -> row (lane<16 ? r : 8+r), col = lane&15
  const int crow = (tm << 4) + (half << 3);
  const int ccol = (tg << 6) + (lane & 15);
  float* cp = C + (size_t)crow * DIM + ccol;
#pragma unroll
  for (int r = 0; r < 8; ++r) {
    cp[(size_t)r * DIM + 0]  = c0[r];
    cp[(size_t)r * DIM + 16] = c1[r];
    cp[(size_t)r * DIM + 32] = c2[r];
    cp[(size_t)r * DIM + 48] = c3[r];
  }
}

// ---------------- graph scatter: agg[tgt] += w_rel * h[src] ----------------
__global__ void k_scatter(const float* __restrict__ h, float* __restrict__ agg,
                          const int* __restrict__ edge_data,
                          const float* __restrict__ r_weight) {
  const int e = blockIdx.x;
  const int t = threadIdx.x;
  const int src = edge_data[e];
  const int rel = edge_data[EFULL + e];
  const int tgt = edge_data[2 * EFULL + e];
  const float w = r_weight[rel];
  const float v = h[(size_t)src * DIM + t] * w;
  atomicAdd(&agg[(size_t)tgt * DIM + t], v);
}

// ---------------- per-column sum / sum-of-squares ----------------
__global__ void k_colstats(const float* __restrict__ x, float* __restrict__ stats) {
  const int c = threadIdx.x;                // 256 columns
  float s = 0.0f, q = 0.0f;
  for (int r = blockIdx.x; r < N_ENT; r += gridDim.x) {
    float v = x[(size_t)r * DIM + c];
    s += v; q += v * v;
  }
  atomicAdd(&stats[c], s);
  atomicAdd(&stats[DIM + c], q);
}

// ---------------- BN + tanh (fused; bf16 output feeds next WMMA GEMM) ------
__global__ void k_bn_tanh_bf16(const float* __restrict__ x, const float* __restrict__ stats,
                               const float* __restrict__ gamma, const float* __restrict__ beta,
                               __bf16* __restrict__ out) {
  const int c = threadIdx.x;
  const float inv_n = 1.0f / (float)N_ENT;
  const float mu = stats[c] * inv_n;
  const float var = stats[DIM + c] * inv_n - mu * mu;
  const float rstd = rsqrtf(var + BN_EPS);
  const float g = gamma[c] * rstd;
  const float b = beta[c] - mu * g;
  for (int r = blockIdx.x; r < N_ENT; r += gridDim.x) {
    float v = x[(size_t)r * DIM + c];
    out[(size_t)r * DIM + c] = (__bf16)tanhf(fmaf(g, v, b));
  }
}

__global__ void k_bn_tanh_f32(const float* __restrict__ x, const float* __restrict__ stats,
                              const float* __restrict__ gamma, const float* __restrict__ beta,
                              float* __restrict__ out) {
  const int c = threadIdx.x;
  const float inv_n = 1.0f / (float)N_ENT;
  const float mu = stats[c] * inv_n;
  const float var = stats[DIM + c] * inv_n - mu * mu;
  const float rstd = rsqrtf(var + BN_EPS);
  const float g = gamma[c] * rstd;
  const float b = beta[c] - mu * g;
  for (int r = blockIdx.x; r < N_ENT; r += gridDim.x) {
    float v = x[(size_t)r * DIM + c];
    out[(size_t)r * DIM + c] = tanhf(fmaf(g, v, b));
  }
}

// ---------------- final edge gather ----------------
__global__ void k_gather(const float* __restrict__ x, const int* __restrict__ ei,
                         float* __restrict__ out0, float* __restrict__ out1) {
  const int e = blockIdx.x;
  const int t = threadIdx.x;
  const int i0 = ei[e];
  const int i1 = ei[EBATCH + e];
  out0[(size_t)e * DIM + t] = x[(size_t)i0 * DIM + t];
  out1[(size_t)e * DIM + t] = x[(size_t)i1 * DIM + t];
}

// ---------------------------------------------------------------------------

extern "C" void kernel_launch(void* const* d_in, const int* in_sizes, int n_in,
                              void* d_out, int out_size, void* d_ws, size_t ws_size,
                              hipStream_t stream) {
  (void)in_sizes; (void)n_in; (void)out_size; (void)ws_size;

  const int*   edge_index = (const int*)d_in[0];     // [2, EBATCH]
  const float* edge_attr  = (const float*)d_in[1];   // [EBATCH, N_REL]
  const int*   edge_data  = (const int*)d_in[2];     // [3, EFULL]
  const float* entity_emb = (const float*)d_in[3];   // [N_ENT, 128]
  const float* r_weight   = (const float*)d_in[4];   // [N_REL, 1]
  const float* gc1_w      = (const float*)d_in[5];   // [128, 256]
  // d_in[6] gc1_b: cancels inside BatchNorm
  const float* gc2_w      = (const float*)d_in[7];   // [256, 256]
  // d_in[8] gc2_b: cancels inside BatchNorm
  const float* r_emb_w    = (const float*)d_in[9];   // [N_REL, 256]
  const float* bn3_g      = (const float*)d_in[10];
  const float* bn3_b      = (const float*)d_in[11];
  const float* bn4_g      = (const float*)d_in[12];
  const float* bn4_b      = (const float*)d_in[13];

  // workspace carve (256B aligned offsets)
  char* ws = (char*)d_ws;
  size_t off = 0;
  auto carve = [&](size_t bytes) -> char* {
    char* p = ws + off;
    off = (off + bytes + 255) & ~(size_t)255;
    return p;
  };
  __bf16* A1    = (__bf16*)carve((size_t)N_ENT * K1 * 2);      // 25.6 MB
  __bf16* W1T   = (__bf16*)carve((size_t)DIM * K1 * 2);
  __bf16* W2T   = (__bf16*)carve((size_t)DIM * DIM * 2);
  __bf16* W3T   = (__bf16*)carve((size_t)DIM * K3PAD * 2);
  float*  H     = (float*) carve((size_t)N_ENT * DIM * 4);     // 102.4 MB (h1, h2, x_final)
  float*  AGG   = (float*) carve((size_t)N_ENT * DIM * 4);     // 102.4 MB
  __bf16* XBF   = (__bf16*)carve((size_t)N_ENT * DIM * 2);     // 51.2 MB
  float*  STATS = (float*) carve((size_t)2 * DIM * 4);
  __bf16* A3    = (__bf16*)AGG;  // alias: AGG is dead by the time A3 is built

  float* out_e0 = (float*)d_out;
  float* out_e1 = out_e0 + (size_t)EBATCH * DIM;
  float* out_ea = out_e1 + (size_t)EBATCH * DIM;

  const int gemm_block = 128;                         // 4 waves/block
  const int wt_ent = (N_ENT / 16) * 4;                // wavetiles, layer GEMMs
  const int wt_eb  = (EBATCH / 16) * 4;               // wavetiles, GEMM3

  // --- stage bf16 operands ---
  k_cvt_pad      <<<4096, 256, 0, stream>>>(entity_emb, A1, N_ENT, K1, K1);
  k_transpose_cvt<<< 256, 256, 0, stream>>>(gc1_w,   W1T, K1,    DIM, K1);
  k_transpose_cvt<<< 256, 256, 0, stream>>>(gc2_w,   W2T, DIM,   DIM, DIM);
  k_transpose_cvt<<< 256, 256, 0, stream>>>(r_emb_w, W3T, N_REL, DIM, K3PAD);

  // --- layer 1: GEMM -> scatter -> BN+tanh (bf16 out) ---
  k_gemm_wmma<<<(wt_ent + 3) / 4, gemm_block, 0, stream>>>(A1, W1T, H, N_ENT, K1);
  k_zero     <<<4096, 256, 0, stream>>>(AGG, (size_t)N_ENT * DIM);
  k_zero     <<<1, 512, 0, stream>>>(STATS, 2 * DIM);
  k_scatter  <<<EFULL, DIM, 0, stream>>>(H, AGG, edge_data, r_weight);
  k_colstats <<<4096, DIM, 0, stream>>>(AGG, STATS);
  k_bn_tanh_bf16<<<8192, DIM, 0, stream>>>(AGG, STATS, bn3_g, bn3_b, XBF);

  // --- layer 2: GEMM -> scatter -> BN+tanh (f32 out into H) ---
  k_gemm_wmma<<<(wt_ent + 3) / 4, gemm_block, 0, stream>>>(XBF, W2T, H, N_ENT, DIM);
  k_zero     <<<4096, 256, 0, stream>>>(AGG, (size_t)N_ENT * DIM);
  k_zero     <<<1, 512, 0, stream>>>(STATS, 2 * DIM);
  k_scatter  <<<EFULL, DIM, 0, stream>>>(H, AGG, edge_data, r_weight);
  k_colstats <<<4096, DIM, 0, stream>>>(AGG, STATS);
  k_bn_tanh_f32<<<8192, DIM, 0, stream>>>(AGG, STATS, bn4_g, bn4_b, H);

  // --- outputs ---
  k_gather   <<<EBATCH, DIM, 0, stream>>>(H, edge_index, out_e0, out_e1);
  k_cvt_pad  <<<4096, 256, 0, stream>>>(edge_attr, A3, EBATCH, N_REL, K3PAD);
  k_gemm_wmma<<<(wt_eb + 3) / 4, gemm_block, 0, stream>>>(A3, W3T, out_ea, EBATCH, K3PAD);
}
